// multi_head_attention_11141145166224
// MI455X (gfx1250) — matrix operand order repaired and checked by reading, not compiled, not run on hardware
//
#include <hip/hip_runtime.h>

typedef __attribute__((ext_vector_type(16))) __bf16          v16bf;
typedef __attribute__((ext_vector_type(16))) unsigned short  v16u;
typedef __attribute__((ext_vector_type(8)))  float           v8f;

#define HEADS 8
#define DM    512
#define BATCH 4
#define SEQ   2048
#define DH    64
#define BHN   (BATCH*HEADS)     // 32
#define NEGINF (-1000000000.0f)

// fp32 -> bf16 via native convert
static __device__ __forceinline__ unsigned short f2bf(float x) {
    __bf16 h = (__bf16)x;
    return __builtin_bit_cast(unsigned short, h);
}

static __device__ __forceinline__ v8f wmma_bf16(v16bf a, v16bf b, v8f c) {
    return __builtin_amdgcn_wmma_f32_16x16x32_bf16(false, a, false, b, (short)0, c, false, false);
}

static __device__ __forceinline__ v16bf ld_frag(const unsigned short* p) {
    return __builtin_bit_cast(v16bf, *(const v16u*)p);     // 2x b128 loads
}

// -------------------------------------------------------------------------
// Kernel 1 (templated on Z): X@W with bf16 WMMA; output written in
// FRAGMENT-READY global layouts for the attention kernel:
//   Z=0: Qa [bh][s/16][frag:2][lane:32][j:16]  (WMMA A-layout, d 0-31/32-63)
//   Z=1: Kf [bh][s/32][frag:4][lane:32][j:16]  (B-layout, frag=(d>>5)*2+(key>>4))
//   Z=2: Vf [bh][s/32][frag:4][lane:32][j:16]  (B-layout, frag=dh>>4) +bias+leaky
// -------------------------------------------------------------------------
template <int Z>
__global__ __launch_bounds__(128) void qkv_proj_kernel(
    const float* __restrict__ X, const float* __restrict__ W,
    const float* __restrict__ bv, unsigned short* __restrict__ dstBuf)
{
    const int rowbase = blockIdx.y * 64;
    const int colbase = blockIdx.x * 64;
    const int tid = threadIdx.x;
    const int w  = tid >> 5;
    const int l  = tid & 31;
    const int lg = l >> 4;
    const int ln = l & 15;

    __shared__ __align__(32) unsigned short sXa[4 * 32 * 16];  // A-layout per wave
    __shared__ __align__(32) unsigned short sWf[4 * 32 * 16];  // B-layout per frag
    __shared__ __align__(32) unsigned short sT[64 * 64];       // epilogue swizzle

    v8f acc[4] = { v8f{}, v8f{}, v8f{}, v8f{} };

    for (int kk = 0; kk < DM; kk += 32) {
        __syncthreads();
        // ---- stage X tile (64 rows x 32 k) directly in A-fragment layout ----
        #pragma unroll
        for (int it = 0; it < 4; ++it) {
            int i  = tid + it * 128;                 // 512 float4s
            int r  = i >> 3;
            int c4 = (i & 7) * 4;                    // k offset, 4 consecutive
            float4 v = *(const float4*)(X + (size_t)(rowbase + r) * DM + kk + c4);
            int wv   = r >> 4;
            int lane = (((c4 >> 3) & 1) << 4) | (r & 15);
            int jb   = ((c4 >> 4) << 3) | (c4 & 7); // 4 consecutive j slots
            uint2 p;
            p.x = (unsigned)f2bf(v.x) | ((unsigned)f2bf(v.y) << 16);
            p.y = (unsigned)f2bf(v.z) | ((unsigned)f2bf(v.w) << 16);
            *(uint2*)(sXa + (wv * 32 + lane) * 16 + jb) = p;   // 8B aligned
        }
        // ---- stage W tile (32 k x 64 cols) in B-fragment layout ----
        #pragma unroll
        for (int it = 0; it < 4; ++it) {
            int i  = tid + it * 128;
            int r  = i >> 4;                         // k 0..31
            int c4 = (i & 15) * 4;                   // col offset
            float4 v = *(const float4*)(W + (size_t)(kk + r) * DM + colbase + c4);
            int f    = c4 >> 4;
            int lane = (((r >> 3) & 1) << 4) | (c4 & 15);
            int j    = ((r >> 4) << 3) | (r & 7);
            sWf[(f * 32 + lane + 0) * 16 + j] = f2bf(v.x);
            sWf[(f * 32 + lane + 1) * 16 + j] = f2bf(v.y);
            sWf[(f * 32 + lane + 2) * 16 + j] = f2bf(v.z);
            sWf[(f * 32 + lane + 3) * 16 + j] = f2bf(v.w);
        }
        __syncthreads();

        v16bf a = ld_frag(sXa + (w * 32 + l) * 16);
        #pragma unroll
        for (int f = 0; f < 4; ++f)
            acc[f] = wmma_bf16(a, ld_frag(sWf + (f * 32 + l) * 16), acc[f]);
    }

    // ---- epilogue: swizzle through LDS, then coalesced dword stores ----
    const int b     = rowbase >> 11;
    const int sbase = rowbase & (SEQ - 1);
    const int h     = colbase >> 6;
    const int bh    = b * HEADS + h;

    float bvv[4];
    if (Z == 2) {
        #pragma unroll
        for (int f = 0; f < 4; ++f) bvv[f] = bv[colbase + f * 16 + ln];
    }

    __syncthreads();
    #pragma unroll
    for (int f = 0; f < 4; ++f) {
        #pragma unroll
        for (int i = 0; i < 8; ++i) {
            int rt = w * 16 + i + 8 * lg;   // tile row (s offset)
            int c  = f * 16 + ln;           // tile col (== dh; colbase%64==0)
            float val = acc[f][i];
            int idx;
            if (Z == 0) {
                int qblk = rt >> 4, frag = c >> 5;
                int lane2 = (((c >> 3) & 1) << 4) | (rt & 15);
                int j2    = (((c >> 4) & 1) << 3) | (c & 7);
                idx = ((qblk * 2 + frag) * 32 + lane2) * 16 + j2;
            } else if (Z == 1) {
                int kb2 = rt >> 5, frag = ((c >> 5) << 1) | ((rt >> 4) & 1);
                int lane2 = (((c >> 3) & 1) << 4) | (rt & 15);
                int j2    = (((c >> 4) & 1) << 3) | (c & 7);
                idx = ((kb2 * 4 + frag) * 32 + lane2) * 16 + j2;
            } else {
                val += bvv[f];
                val = val > 0.0f ? val : 0.01f * val;   // leaky_relu
                int kb2 = rt >> 5, frag = c >> 4;
                int lane2 = (((rt >> 3) & 1) << 4) | (c & 15);
                int j2    = (((rt >> 4) & 1) << 3) | (rt & 7);
                idx = ((kb2 * 4 + frag) * 32 + lane2) * 16 + j2;
            }
            sT[idx] = f2bf(val);
        }
    }
    __syncthreads();

    unsigned* dst;
    if (Z == 0) dst = (unsigned*)(dstBuf + ((size_t)bh * (SEQ / 16) + (sbase >> 4)) * 1024);
    else        dst = (unsigned*)(dstBuf + ((size_t)bh * (SEQ / 32) + (sbase >> 5)) * 2048);
    const unsigned* src = (const unsigned*)sT;
    #pragma unroll
    for (int it = 0; it < 16; ++it) dst[tid + it * 128] = src[tid + it * 128];
}

// -------------------------------------------------------------------------
// Kernel 2: flash attention. 4 independent waves x 16 queries; all WMMA
// operands loaded as b128 vectors from fragment-ready L2-resident buffers.
// Diagonal mask applied only in the single overlapping key block (wave-
// uniform scalar branch); per-lane partial row sums reduced once at the end.
// -------------------------------------------------------------------------
__global__ __launch_bounds__(128) void flash_attn_kernel(
    const unsigned short* __restrict__ qa, const unsigned short* __restrict__ kf,
    const unsigned short* __restrict__ vf, float* __restrict__ outCtx,
    float* __restrict__ partial)
{
    const int bh = blockIdx.y;
    const int qt = blockIdx.x;
    const int tid = threadIdx.x;
    const int w  = tid >> 5;
    const int l  = tid & 31;
    const int lg = l >> 4;
    const int ln = l & 15;
    const int qbase = qt * 64;
    const int qw    = qbase + w * 16;        // this wave's first query row

    __shared__ __align__(32) unsigned short sPa[4][512];   // per-wave P, A-layout
    __shared__ float sRed[4];

    // Q A-fragments
    const unsigned short* qp =
        qa + ((size_t)bh * (SEQ / 16) + (qbase >> 4) + w) * 1024;
    const v16bf aQ0 = ld_frag(qp + l * 16);
    const v16bf aQ1 = ld_frag(qp + 512 + l * 16);

    v8f acc[4] = { v8f{}, v8f{}, v8f{}, v8f{} };
    float m[8], lsum[8];                    // lsum: per-LANE partial row sum
    #pragma unroll
    for (int i = 0; i < 8; ++i) { m[i] = -INFINITY; lsum[i] = 0.0f; }
    float reg2 = 0.0f;

    const unsigned short* kB = kf + (size_t)bh * (SEQ / 32) * 2048;
    const unsigned short* vB = vf + (size_t)bh * (SEQ / 32) * 2048;

    for (int kb = 0; kb < SEQ; kb += 32) {
        const unsigned short* kblk = kB + (size_t)(kb >> 5) * 2048;
        const unsigned short* vblk = vB + (size_t)(kb >> 5) * 2048;
        if (kb + 32 < SEQ) {   // global_prefetch_b8 next tiles
            __builtin_prefetch(kblk + 2048 + (size_t)l * 64, 0, 1);
            __builtin_prefetch(vblk + 2048 + (size_t)l * 64, 0, 1);
        }

        // S = Q K^T : 16 queries x 32 keys, Dh=64 contraction
        float st[2][8];
        #pragma unroll
        for (int nh = 0; nh < 2; ++nh) {
            v16bf b0 = ld_frag(kblk + (0 * 2 + nh) * 512 + l * 16);
            v16bf b1 = ld_frag(kblk + (1 * 2 + nh) * 512 + l * 16);
            v8f s8 = {};
            s8 = wmma_bf16(aQ0, b0, s8);
            s8 = wmma_bf16(aQ1, b1, s8);
            #pragma unroll
            for (int i = 0; i < 8; ++i) {
                float sv = s8[i];
                reg2 += sv * sv;                 // raw, pre-scale/mask
                st[nh][i] = sv * 0.125f;         // dh^-0.5
            }
        }

        // Diagonal self-mask: only the single 32-aligned block containing qw
        // can intersect this wave's 16 query rows (wave-uniform branch).
        if (kb == (qw & ~31)) {
            #pragma unroll
            for (int nh = 0; nh < 2; ++nh) {
                int kpos = kb + nh * 16 + ln;
                #pragma unroll
                for (int i = 0; i < 8; ++i) {
                    int qpos = qw + i + 8 * lg;
                    if (qpos == kpos) st[nh][i] = NEGINF;
                }
            }
        }

        // Online softmax; write P directly in A-fragment layout
        #pragma unroll
        for (int i = 0; i < 8; ++i) {
            float t = fmaxf(st[0][i], st[1][i]);
            t = fmaxf(t, __shfl_xor(t, 1, 32));
            t = fmaxf(t, __shfl_xor(t, 2, 32));
            t = fmaxf(t, __shfl_xor(t, 4, 32));
            t = fmaxf(t, __shfl_xor(t, 8, 32));
            float mnew = fmaxf(m[i], t);
            float p0 = __expf(st[0][i] - mnew);
            float p1 = __expf(st[1][i] - mnew);
            float so = __expf(m[i] - mnew);      // row-uniform rescale
            lsum[i] = lsum[i] * so + (p0 + p1);  // per-lane partial; reduce later
            m[i] = mnew;
            acc[0][i] *= so; acc[1][i] *= so; acc[2][i] *= so; acc[3][i] *= so;
            // A-layout target: lane' = ((col>>3)&1)*16 + row ; j = col&7 (+8 for p1)
            int base = ((((ln >> 3) & 1) << 4) | (i + 8 * lg)) * 16 + (ln & 7);
            sPa[w][base]     = f2bf(p0);
            sPa[w][base + 8] = f2bf(p1);
        }
        // wave-private LDS is in-order; pin compiler + HW ordering
        asm volatile("s_wait_dscnt 0" ::: "memory");

        v16bf aP = ld_frag(&sPa[w][l * 16]);
        #pragma unroll
        for (int f = 0; f < 4; ++f)
            acc[f] = wmma_bf16(aP, ld_frag(vblk + f * 512 + l * 16), acc[f]);
    }

    // Merged-head context [B][S][D], fp32 (reduce per-lane lsum partials once)
    const int b = bh >> 3, h = bh & 7;
    #pragma unroll
    for (int i = 0; i < 8; ++i) {
        float s = lsum[i];
        s += __shfl_xor(s, 1, 32);
        s += __shfl_xor(s, 2, 32);
        s += __shfl_xor(s, 4, 32);
        s += __shfl_xor(s, 8, 32);
        float inv = 1.0f / s;
        int qpos = qbase + w * 16 + i + 8 * lg;
        float* orow = outCtx + ((size_t)b * SEQ + qpos) * DM + h * DH;
        #pragma unroll
        for (int f = 0; f < 4; ++f)
            orow[f * 16 + ln] = acc[f][i] * inv;
    }

    // Deterministic per-block sum(s^2) partial
    float r = reg2;
    r += __shfl_xor(r, 1, 32);
    r += __shfl_xor(r, 2, 32);
    r += __shfl_xor(r, 4, 32);
    r += __shfl_xor(r, 8, 32);
    r += __shfl_xor(r, 16, 32);
    if (l == 0) sRed[w] = r;
    __syncthreads();
    if (tid == 0)
        partial[blockIdx.y * gridDim.x + blockIdx.x] = sRed[0] + sRed[1] + sRed[2] + sRed[3];
}

// -------------------------------------------------------------------------
// Kernel 3: fixed-order reduction of partials -> reg_atten (bit-stable)
// -------------------------------------------------------------------------
__global__ __launch_bounds__(128) void reg_reduce_kernel(
    const float* __restrict__ partial, float* __restrict__ outReg, int n)
{
    __shared__ double sd[128];
    double acc = 0.0;
    for (int i = threadIdx.x; i < n; i += 128) acc += (double)partial[i];
    sd[threadIdx.x] = acc;
    __syncthreads();
    for (int s = 64; s > 0; s >>= 1) {
        if ((int)threadIdx.x < s) sd[threadIdx.x] += sd[threadIdx.x + s];
        __syncthreads();
    }
    if (threadIdx.x == 0)
        *outReg = (float)(sd[0] / ((double)BHN * (double)SEQ * (double)SEQ));
}

// -------------------------------------------------------------------------
extern "C" void kernel_launch(void* const* d_in, const int* in_sizes, int n_in,
                              void* d_out, int out_size, void* d_ws, size_t ws_size,
                              hipStream_t stream) {
    (void)in_sizes; (void)n_in; (void)out_size; (void)ws_size;
    const float* sa = (const float*)d_in[0];
    const float* si = (const float*)d_in[1];
    const float* Wq = (const float*)d_in[2];
    const float* Wk = (const float*)d_in[3];
    const float* Wv = (const float*)d_in[4];
    const float* bv = (const float*)d_in[5];

    float* out    = (float*)d_out;
    float* outCtx = out;
    float* outReg = out + (size_t)BATCH * SEQ * DM;

    char* ws = (char*)d_ws;
    const size_t hsz = (size_t)BHN * SEQ * DH * sizeof(unsigned short);  // 8 MB each
    unsigned short* qa = (unsigned short*)(ws);
    unsigned short* kf = (unsigned short*)(ws + hsz);
    unsigned short* vf = (unsigned short*)(ws + 2 * hsz);
    float* partial     = (float*)(ws + 3 * hsz);

    const dim3 g1(DM / 64, (BATCH * SEQ) / 64);
    qkv_proj_kernel<0><<<g1, 128, 0, stream>>>(si, Wq, bv, qa);
    qkv_proj_kernel<1><<<g1, 128, 0, stream>>>(sa, Wk, bv, kf);
    qkv_proj_kernel<2><<<g1, 128, 0, stream>>>(sa, Wv, bv, vf);

    flash_attn_kernel<<<dim3(SEQ / 64, BHN), 128, 0, stream>>>(
        qa, kf, vf, outCtx, partial);

    reg_reduce_kernel<<<1, 128, 0, stream>>>(partial, outReg, (SEQ / 64) * BHN);
}
